// VeduBitLinear_44135083934122
// MI455X (gfx1250) — matrix operand order repaired and checked by reading, not compiled, not run on hardware
//
#include <hip/hip_runtime.h>

typedef _Float16 half_t;
typedef _Float16 v4h  __attribute__((ext_vector_type(4)));
typedef _Float16 v8h  __attribute__((ext_vector_type(8)));
typedef _Float16 v16h __attribute__((ext_vector_type(16)));
typedef float    v4f  __attribute__((ext_vector_type(4)));
typedef float    v8f  __attribute__((ext_vector_type(8)));
typedef int      v4i  __attribute__((ext_vector_type(4)));

union V16 { v16h v; v8h h[2]; };

constexpr int   D_IN   = 2048;
constexpr int   D_OUT  = 2048;
constexpr float LN_EPS = 1e-5f;
constexpr float Q_EPS  = 1e-5f;

// ---------------- CDNA5 async global->LDS helpers ----------------
#if __has_builtin(__builtin_amdgcn_global_load_async_to_lds_b128)
#define ASYNC_BUILTIN 1
#endif

typedef __attribute__((address_space(1))) v4i* gv4i_p;   // global int4*
typedef __attribute__((address_space(3))) v4i* lv4i_p;   // LDS int4*

__device__ __forceinline__ void async_b128(const half_t* g, half_t* l) {
#ifdef ASYNC_BUILTIN
    __builtin_amdgcn_global_load_async_to_lds_b128(
        (gv4i_p)(v4i*)g, (lv4i_p)(v4i*)l, 0, 0);
#else
    unsigned loff = (unsigned)(unsigned long long)l;   // LDS aperture: addr[31:0] = LDS offset
    asm volatile("global_load_async_to_lds_b128 %0, %1, off"
                 :: "v"(loff), "v"(g) : "memory");
#endif
}

template <int N>
__device__ __forceinline__ void wait_asynccnt() {
#if __has_builtin(__builtin_amdgcn_s_wait_asynccnt)
    __builtin_amdgcn_s_wait_asynccnt(N);
#else
    asm volatile("s_wait_asynccnt %0" :: "n"(N) : "memory");
#endif
}

// ---------------- Kernel 1a: partial sums of |W| (deterministic 2-pass) ----------------
__global__ void k_abs_part(const float* __restrict__ w, float* __restrict__ partials, int n4) {
    __shared__ float red[256];
    float s = 0.f;
    const v4f* w4 = reinterpret_cast<const v4f*>(w);
    for (int i = blockIdx.x * blockDim.x + threadIdx.x; i < n4; i += gridDim.x * blockDim.x) {
        v4f v = w4[i];
        s += fabsf(v[0]) + fabsf(v[1]) + fabsf(v[2]) + fabsf(v[3]);
    }
    red[threadIdx.x] = s;
    __syncthreads();
    for (int off = 128; off > 0; off >>= 1) {
        if ((int)threadIdx.x < off) red[threadIdx.x] += red[threadIdx.x + off];
        __syncthreads();
    }
    if (threadIdx.x == 0) partials[blockIdx.x] = red[0];
}

// ---------------- Kernel 1b: final deterministic reduce (256 partials) ----------------
__global__ void k_abs_final(const float* __restrict__ partials, float* __restrict__ sum_out) {
    __shared__ float red[256];
    red[threadIdx.x] = partials[threadIdx.x];
    __syncthreads();
    for (int off = 128; off > 0; off >>= 1) {
        if ((int)threadIdx.x < off) red[threadIdx.x] += red[threadIdx.x + off];
        __syncthreads();
    }
    if (threadIdx.x == 0) sum_out[0] = red[0];
}

// ---------------- Kernel 2: ternary quantize -> f16 {-1,0,1} ----------------
__global__ void k_quant(const float* __restrict__ w, const float* __restrict__ sum_abs,
                        half_t* __restrict__ wq, int n4) {
    int i = blockIdx.x * blockDim.x + threadIdx.x;
    if (i >= n4) return;
    const float scale = sum_abs[0] * (1.0f / (float)((size_t)D_IN * D_OUT));
    const float inv   = 1.0f / (scale + Q_EPS);
    v4f v = reinterpret_cast<const v4f*>(w)[i];
    v4h q;
#pragma unroll
    for (int k = 0; k < 4; ++k) {
        float t = rintf(v[k] * inv);
        t = fminf(1.0f, fmaxf(-1.0f, t));
        q[k] = (half_t)t;
    }
    reinterpret_cast<v4h*>(wq)[i] = q;
}

// ---------------- Kernel 3: LayerNorm row -> f16 ----------------
__global__ void k_layernorm(const float* __restrict__ x, const float* __restrict__ gamma,
                            const float* __restrict__ beta, half_t* __restrict__ xn) {
    __shared__ float s1[256];
    __shared__ float s2[256];
    const int row = blockIdx.x;
    const int t   = threadIdx.x;
    const float* xr = x + (size_t)row * D_IN;
    v4f a = reinterpret_cast<const v4f*>(xr)[2 * t + 0];
    v4f b = reinterpret_cast<const v4f*>(xr)[2 * t + 1];
    float s = 0.f, sq = 0.f;
#pragma unroll
    for (int k = 0; k < 4; ++k) { s += a[k]; sq += a[k] * a[k]; }
#pragma unroll
    for (int k = 0; k < 4; ++k) { s += b[k]; sq += b[k] * b[k]; }
    s1[t] = s; s2[t] = sq;
    __syncthreads();
    for (int off = 128; off > 0; off >>= 1) {
        if (t < off) { s1[t] += s1[t + off]; s2[t] += s2[t + off]; }
        __syncthreads();
    }
    const float mean = s1[0] * (1.0f / (float)D_IN);
    const float var  = s2[0] * (1.0f / (float)D_IN) - mean * mean;
    const float rstd = rsqrtf(var + LN_EPS);
    v4f g0 = reinterpret_cast<const v4f*>(gamma)[2 * t + 0];
    v4f g1 = reinterpret_cast<const v4f*>(gamma)[2 * t + 1];
    v4f b0 = reinterpret_cast<const v4f*>(beta)[2 * t + 0];
    v4f b1 = reinterpret_cast<const v4f*>(beta)[2 * t + 1];
    v8h o;
#pragma unroll
    for (int k = 0; k < 4; ++k) o[k]     = (half_t)((a[k] - mean) * rstd * g0[k] + b0[k]);
#pragma unroll
    for (int k = 0; k < 4; ++k) o[4 + k] = (half_t)((b[k] - mean) * rstd * g1[k] + b1[k]);
    reinterpret_cast<v8h*>(xn + (size_t)row * D_IN)[t] = o;
}

// ---------------- Kernel 4: WMMA GEMM  out[M,N] = Xn[M,K] * Wq[N,K]^T + bias ----------------
constexpr int BM     = 128;
constexpr int BN     = 128;
constexpr int KC     = 64;          // halves per LDS stage (2 WMMA k-steps)
constexpr int LDT    = 72;          // LDS row pitch in halves (64 + 8 pad -> conflict-free b128)
constexpr int TILE   = 128 * LDT;   // halves per matrix per stage
constexpr int NSTAGE = 3;           // async ring depth

__global__ __launch_bounds__(256)
void k_gemm(const half_t* __restrict__ A,   // [M, K] activations f16
            const half_t* __restrict__ Bw,  // [N, K] ternary weights f16
            const float* __restrict__ bias, // [N]
            float* __restrict__ out,        // [M, N]
            int M) {
    __shared__ half_t lds[NSTAGE * 2 * TILE];   // per stage: A[128][LDT], B[128][LDT]

    const int t    = threadIdx.x;
    const int lane = t & 31;
    const int wid  = t >> 5;      // 0..7
    const int wm   = wid >> 1;    // 0..3 : 32 output rows each
    const int wn   = wid & 1;     // 0..1 : 64 output cols each
    const int m0   = blockIdx.y * BM;
    const int n0   = blockIdx.x * BN;

    // global->LDS staging: thread owns 32 contiguous halves of one row per stage
    const int lrow = t >> 1;              // 0..127
    const int lk   = (t & 1) * 32;        // 0 or 32
    const half_t* gA = A  + (size_t)(m0 + lrow) * D_IN + lk;
    const half_t* gB = Bw + (size_t)(n0 + lrow) * D_IN + lk;
    const int soff = lrow * LDT + lk;

    // issue one stage: 8 async b128 DMA copies (ASYNCcnt += 8)
    auto issue = [&](int s, int b) {
        half_t* la = lds + b * (2 * TILE) + soff;
        half_t* lb = la + TILE;
        const half_t* pa = gA + (size_t)s * KC;
        const half_t* pb = gB + (size_t)s * KC;
#pragma unroll
        for (int i = 0; i < 4; ++i) {
            async_b128(pa + 8 * i, la + 8 * i);
            async_b128(pb + 8 * i, lb + 8 * i);
        }
    };

    issue(0, 0);
    issue(1, 1);

    v8f acc[2][4];
#pragma unroll
    for (int i = 0; i < 2; ++i)
#pragma unroll
        for (int j = 0; j < 4; ++j)
#pragma unroll
            for (int r = 0; r < 8; ++r) acc[i][j][r] = 0.f;

    const int mlane = lane & 15;
    const int khiA  = 8  * (lane >> 4);   // A frag: lanes 16-31 start at K+8
    const int khiB  = 8  * (lane >> 4);   // B frag: lanes 16-31 start at K+8

    const int NKB = D_IN / KC;            // 32 stages
    for (int kb = 0; kb < NKB; ++kb) {
        // my stage-kb DMAs complete (in-order): <=8 outstanding means only stage kb+1 left
        if (kb + 1 < NKB) wait_asynccnt<8>();
        else              wait_asynccnt<0>();
        __syncthreads();                  // everyone's stage-kb tile is in LDS

        if (kb + 2 < NKB) issue(kb + 2, (kb + 2) % 3);   // overlap DMA with compute

        const half_t* Ab = lds + (kb % 3) * (2 * TILE);
        const half_t* Bb = Ab + TILE;
#pragma unroll
        for (int kk = 0; kk < KC; kk += 32) {
            V16 af[2], bf[4];
#pragma unroll
            for (int i = 0; i < 2; ++i) {
                const half_t* p = Ab + (wm * 32 + i * 16 + mlane) * LDT + kk;
                af[i].h[0] = *reinterpret_cast<const v8h*>(p + khiA);
                af[i].h[1] = *reinterpret_cast<const v8h*>(p + 16 + khiA);
            }
#pragma unroll
            for (int j = 0; j < 4; ++j) {
                const half_t* p = Bb + (wn * 64 + j * 16 + mlane) * LDT + kk + khiB;
                bf[j].h[0] = *reinterpret_cast<const v8h*>(p);
                bf[j].h[1] = *reinterpret_cast<const v8h*>(p + 16);
            }
#pragma unroll
            for (int i = 0; i < 2; ++i)
#pragma unroll
                for (int j = 0; j < 4; ++j)
                    acc[i][j] = __builtin_amdgcn_wmma_f32_16x16x32_f16(
                        false, af[i].v, false, bf[j].v,
                        (short)0, acc[i][j], false, false);
        }
    }

    // Epilogue: C/D layout — VGPR r: lanes 0-15 -> M=r, lanes 16-31 -> M=r+8; N = lane&15
#pragma unroll
    for (int i = 0; i < 2; ++i) {
        const int rbase = m0 + wm * 32 + i * 16 + (lane >> 4) * 8;
#pragma unroll
        for (int j = 0; j < 4; ++j) {
            const int col = n0 + wn * 64 + j * 16 + mlane;
            const float bv = bias[col];
#pragma unroll
            for (int r = 0; r < 8; ++r)
                out[(size_t)(rbase + r) * D_OUT + col] = acc[i][j][r] + bv;
        }
    }
}

// ---------------- Launch ----------------
extern "C" void kernel_launch(void* const* d_in, const int* in_sizes, int n_in,
                              void* d_out, int out_size, void* d_ws, size_t ws_size,
                              hipStream_t stream) {
    const float* x     = (const float*)d_in[0];
    const float* w     = (const float*)d_in[1];
    const float* bias  = (const float*)d_in[2];
    const float* gamma = (const float*)d_in[3];
    const float* beta  = (const float*)d_in[4];
    float* out = (float*)d_out;

    char* ws = (char*)d_ws;
    float*  sum_final = (float*)ws;                 // 4 B
    float*  partials  = (float*)(ws + 1024);        // 256 floats
    half_t* wq        = (half_t*)(ws + 4096);       // 8 MiB
    half_t* xn        = (half_t*)(ws + 4096 + (size_t)D_OUT * D_IN * sizeof(half_t)); // 32 MiB

    const int M  = in_sizes[0] / D_IN;              // 8192
    const int NW = D_IN * D_OUT;                    // 4194304

    k_abs_part<<<256, 256, 0, stream>>>(w, partials, NW / 4);
    k_abs_final<<<1, 256, 0, stream>>>(partials, sum_final);
    k_quant<<<(NW / 4 + 255) / 256, 256, 0, stream>>>(w, sum_final, wq, NW / 4);
    k_layernorm<<<M, 256, 0, stream>>>(x, gamma, beta, xn);

    dim3 grid(D_OUT / BN, M / BM);                  // (16, 64)
    k_gemm<<<grid, 256, 0, stream>>>(xn, wq, bias, out, M);
}
